// GCN_15161234555392
// MI455X (gfx1250) — matrix operand order repaired and checked by reading, not compiled, not run on hardware
//
#include <hip/hip_runtime.h>
#include <hip/hip_bf16.h>

typedef float v2f  __attribute__((ext_vector_type(2)));
typedef float v8f  __attribute__((ext_vector_type(8)));
typedef _Float16 v16h __attribute__((ext_vector_type(16)));
typedef unsigned int v4u __attribute__((ext_vector_type(4)));
typedef int v4i_t __attribute__((ext_vector_type(4)));
typedef int v8i_t __attribute__((ext_vector_type(8)));

#if __has_builtin(__builtin_amdgcn_wmma_f32_16x16x4_f32)
#define USE_F32_WMMA 1
#else
#define USE_F32_WMMA 0
#endif

#if __has_builtin(__builtin_amdgcn_tensor_load_to_lds)
#define USE_TDM 1
#if __has_include(<hip/amd_detail/amd_gfx1250_TDM.h>)
#define TDM_ARGS6 1  // therock-10.0 headers -> 6-arg builtin
#else
#define TDM_ARGS6 0  // ROCm 7.2 -> 5-arg builtin
#endif
#else
#define USE_TDM 0
#endif

// ==================== CSR construction (counting sort by dst) ====================

__global__ void k_count(const long long* __restrict__ dst, int* __restrict__ cnt, int e) {
  int i = blockIdx.x * blockDim.x + threadIdx.x;
  if (i < e) atomicAdd(&cnt[(int)dst[i]], 1);
}

// phase 1: per-1024-chunk totals
__global__ void k_scan_partial(const int* __restrict__ cnt, int* __restrict__ blockSums, int n) {
  __shared__ int sm[256];
  const int b = blockIdx.x, tid = threadIdx.x;
  const int base = b * 1024 + tid * 4;
  int s = 0;
#pragma unroll
  for (int q = 0; q < 4; ++q) {
    const int idx = base + q;
    if (idx < n) s += cnt[idx];
  }
  sm[tid] = s;
  __syncthreads();
  for (int off = 128; off > 0; off >>= 1) {
    if (tid < off) sm[tid] += sm[tid + off];
    __syncthreads();
  }
  if (tid == 0) blockSums[b] = sm[0];
}

// phase 2: exclusive scan of chunk totals (G <= 128), writes rowptr[n] = total
__global__ void k_scan_bases(int* __restrict__ blockSums, int* __restrict__ rowptr, int G, int n) {
  __shared__ int sm[128];
  const int tid = threadIdx.x;  // block of 128
  const int v0 = (tid < G) ? blockSums[tid] : 0;
  sm[tid] = v0;
  __syncthreads();
  for (int off = 1; off < 128; off <<= 1) {
    int add = (tid >= off) ? sm[tid - off] : 0;
    __syncthreads();
    sm[tid] += add;
    __syncthreads();
  }
  blockSums[tid] = sm[tid] - v0;  // exclusive base per chunk
  if (tid == 127) rowptr[n] = sm[127];
}

// phase 3: per-chunk exclusive scan + chunk base -> rowptr[0..n)
__global__ void k_scan_final(const int* __restrict__ cnt, const int* __restrict__ blockSums,
                             int* __restrict__ rowptr, int n) {
  __shared__ int sm[256];
  const int b = blockIdx.x, tid = threadIdx.x;
  const int base = b * 1024 + tid * 4;
  int v[4];
  int s = 0;
#pragma unroll
  for (int q = 0; q < 4; ++q) {
    const int idx = base + q;
    v[q] = (idx < n) ? cnt[idx] : 0;
    s += v[q];
  }
  sm[tid] = s;
  __syncthreads();
  for (int off = 1; off < 256; off <<= 1) {
    int add = (tid >= off) ? sm[tid - off] : 0;
    __syncthreads();
    sm[tid] += add;
    __syncthreads();
  }
  int run = blockSums[b] + sm[tid] - s;
#pragma unroll
  for (int q = 0; q < 4; ++q) {
    const int idx = base + q;
    if (idx < n) rowptr[idx] = run;
    run += v[q];
  }
}

__global__ void k_fill(const long long* __restrict__ src, const long long* __restrict__ dst,
                       int* __restrict__ cursor, int* __restrict__ sortedSrc, int e) {
  int i = blockIdx.x * blockDim.x + threadIdx.x;
  if (i < e) {
    const int d = (int)dst[i];
    const int slot = atomicAdd(&cursor[d], 1);
    sortedSrc[slot] = (int)src[i];
  }
}

__global__ void k_dinv(const int* __restrict__ rowptr, float* __restrict__ dinv, int n) {
  int i = blockIdx.x * blockDim.x + threadIdx.x;
  if (i < n) dinv[i] = rsqrtf((float)(rowptr[i + 1] - rowptr[i] + 1));  // +1 self-loop
}

// ==================== WMMA GEMM: Hs = (X @ W) * dinv[row] ====================
// Workgroup = 8 waves = 8 M-tiles x one 64-col strip. W staged in LDS in 64-row
// K-chunks via TDM tensor_load_to_lds (pad feature reproduces the 80-float padded
// row stride -> the two B rows hit disjoint bank groups); cooperative-load fallback.
// Requires M%16==0, K%64==0, NC%64==0 (here 100000 / 256 / {256,64}).

#define WK 64
#define WPAD 80

#if USE_TDM
// One TDM 2D tile load: rows x 64 fp32 elements, row stride NC elements in memory,
// padded in LDS: 64 dwords stored (pad_interval=5 -> 2^(5+1)=64) then 16 dwords
// skipped (pad_amount=15 -> 16) => 80-float LDS row stride.
__device__ __forceinline__ void tdm_load_tile(const float* gptr, unsigned lds_off,
                                              unsigned nc_elems, unsigned rows) {
  const unsigned long long ga = reinterpret_cast<unsigned long long>(gptr);
  v4u g0;
  g0[0] = 1u;                                                 // count=1, user descriptor
  g0[1] = lds_off;                                            // lds_addr (bytes)
  g0[2] = (unsigned)ga;                                       // global_addr[31:0]
  g0[3] = (unsigned)((ga >> 32) & 0x01FFFFFFu) | (2u << 30);  // global_addr[56:32]|type=2
  v8i_t g1;
  g1[0] = (int)((2u << 16)      // data_size = 4 bytes
                | (1u << 20)    // pad_enable
                | (5u << 22)    // pad_interval: 64 dwords
                | (15u << 25)); // pad_amount: 16 dwords
  g1[1] = (int)((nc_elems & 0xFFFFu) << 16);                          // tensor_dim0[15:0]
  g1[2] = (int)(((nc_elems >> 16) & 0xFFFFu) | ((rows & 0xFFFFu) << 16));  // dim0 hi|dim1 lo
  g1[3] = (int)(((rows >> 16) & 0xFFFFu) | (64u << 16));              // dim1 hi | tile_dim0=64
  g1[4] = (int)(rows & 0xFFFFu);                                      // tile_dim1 | tile_dim2=0
  g1[5] = (int)nc_elems;                                              // tensor_dim0_stride lo
  g1[6] = 0;
  g1[7] = 0;
  v4i_t z4 = {0, 0, 0, 0};
#if TDM_ARGS6
  v8i_t z8 = {0, 0, 0, 0, 0, 0, 0, 0};
  __builtin_amdgcn_tensor_load_to_lds(g0, g1, z4, z4, z8, 0);
#else
  __builtin_amdgcn_tensor_load_to_lds(g0, g1, z4, z4, 0);
#endif
  __builtin_amdgcn_s_wait_tensorcnt(0);
}
#endif

__global__ void k_gemm_scale(const float* __restrict__ X, const float* __restrict__ W,
                             const float* __restrict__ dinv, float* __restrict__ Hs,
                             int M, int K, int NC) {
  __shared__ float Wl[WK * WPAD];  // 20 KB
  const int strips = NC >> 6;
  const int mtiles = M >> 4;
  const int strip = (int)(blockIdx.x % strips);
  const int mblock = (int)(blockIdx.x / strips);
  const int waveId = threadIdx.x >> 5;
  const int lane = threadIdx.x & 31;
  const int mt = mblock * 8 + waveId;
  const bool active = (mt < mtiles);  // wave-uniform
  const int col0 = strip << 6;
  const int lo = lane & 15;
  const int half = lane >> 4;
  const int row0 = mt << 4;
  const float* __restrict__ xrow = X + (size_t)(row0 + lo) * (size_t)K;

  v8f c[4] = {};
  for (int kc = 0; kc < K; kc += WK) {
    __syncthreads();  // previous chunk's compute done before overwriting LDS
#if USE_TDM
    if (waveId == 0) {  // one DMA per workgroup; TENSORcnt is per-wave
      tdm_load_tile(W + (size_t)kc * (size_t)NC + col0,
                    (unsigned)reinterpret_cast<unsigned long long>(&Wl[0]),
                    (unsigned)NC, (unsigned)WK);
    }
#else
    {
      const int tid = threadIdx.x;
#pragma unroll
      for (int it = 0; it < (WK * 16) / 256; ++it) {
        const int idx = it * 256 + tid;
        const int r = idx >> 4;
        const int sg = (idx & 15) << 2;
        const float4 w4 = *(const float4*)(W + (size_t)(kc + r) * (size_t)NC + col0 + sg);
        *(float4*)(&Wl[r * WPAD + sg]) = w4;
      }
    }
#endif
    __syncthreads();
    if (active) {
#if USE_F32_WMMA
#pragma unroll 4
      for (int k0 = kc; k0 < kc + WK; k0 += 4) {
        const int ka = k0 + (half << 1);
        const float2 av = *(const float2*)(xrow + ka);
        v2f a;
        a.x = av.x;
        a.y = av.y;
        const float* wl = &Wl[(ka - kc) * WPAD + lo];
#pragma unroll
        for (int t = 0; t < 4; ++t) {
          v2f b;
          b.x = wl[t * 16];
          b.y = wl[t * 16 + WPAD];
          c[t] = __builtin_amdgcn_wmma_f32_16x16x4_f32(false, a, false, b, (short)0, c[t],
                                                       false, false);
        }
      }
#else
      for (int k0 = kc; k0 < kc + WK; k0 += 32) {
        v16h a;
        const int a1 = k0 + (half << 4);
#pragma unroll
        for (int p = 0; p < 8; ++p) a[p] = (_Float16)xrow[a1 + p];
#pragma unroll
        for (int p = 0; p < 8; ++p) a[8 + p] = (_Float16)xrow[a1 + 8 + p];
#pragma unroll
        for (int t = 0; t < 4; ++t) {
          v16h b;
          const float* wc = &Wl[(k0 - kc + (half << 4)) * WPAD + (t << 4) + lo];
#pragma unroll
          for (int p = 0; p < 16; ++p) b[p] = (_Float16)wc[p * WPAD];
          c[t] = __builtin_amdgcn_wmma_f32_16x16x32_f16(false, a, false, b, (short)0, c[t],
                                                        false, false);
        }
      }
#endif
    }
  }
  if (active) {
    float rs[8];
#pragma unroll
    for (int v = 0; v < 8; ++v) rs[v] = dinv[row0 + (half << 3) + v];
#pragma unroll
    for (int t = 0; t < 4; ++t) {
#pragma unroll
      for (int v = 0; v < 8; ++v) {
        const int row = row0 + (half << 3) + v;
        Hs[(size_t)row * (size_t)NC + col0 + (t << 4) + lo] = c[t][v] * rs[v];
      }
    }
  }
}

// ==================== layer-1 aggregation: gather-sum + bias + ReLU ====================
// One 64-thread group per node (4 nodes / 256-thread block), float4 per thread:
// each edge is a 1 KB coalesced gather; zero atomics. Edge loop unrolled x2.

__global__ void k_agg_relu(const float* __restrict__ Hs, const int* __restrict__ rowptr,
                           const int* __restrict__ srcidx, const float* __restrict__ dinv,
                           const float* __restrict__ bias, float* __restrict__ h, int N) {
  const int grp = threadIdx.x >> 6;
  const int i = blockIdx.x * 4 + grp;
  if (i >= N) return;  // uniform per wave
  const int c4 = (threadIdx.x & 63) << 2;
  const int beg = rowptr[i], end = rowptr[i + 1];
  float4 acc = *(const float4*)(Hs + (size_t)i * 256 + c4);  // self-loop term
  int e = beg;
  for (; e + 2 <= end; e += 2) {
    const int s0 = srcidx[e];
    const int s1 = srcidx[e + 1];
    const float4 v0 = *(const float4*)(Hs + (size_t)s0 * 256 + c4);
    const float4 v1 = *(const float4*)(Hs + (size_t)s1 * 256 + c4);
    acc.x += v0.x + v1.x;
    acc.y += v0.y + v1.y;
    acc.z += v0.z + v1.z;
    acc.w += v0.w + v1.w;
  }
  if (e < end) {
    const int s = srcidx[e];
    const float4 v = *(const float4*)(Hs + (size_t)s * 256 + c4);
    acc.x += v.x;
    acc.y += v.y;
    acc.z += v.z;
    acc.w += v.w;
  }
  const float di = dinv[i];
  const float4 bb = *(const float4*)(bias + c4);
  float4 o;
  o.x = fmaxf(acc.x * di + bb.x, 0.0f);
  o.y = fmaxf(acc.y * di + bb.y, 0.0f);
  o.z = fmaxf(acc.z * di + bb.z, 0.0f);
  o.w = fmaxf(acc.w * di + bb.w, 0.0f);
  *(float4*)(h + (size_t)i * 256 + c4) = o;
}

// ======== layer-2 aggregation + bias + log_softmax (one wave32 per node, 64 cols) ========

__global__ void k_agg_logsoftmax(const float* __restrict__ Hs, const int* __restrict__ rowptr,
                                 const int* __restrict__ srcidx, const float* __restrict__ dinv,
                                 const float* __restrict__ bias, float* __restrict__ out, int n) {
  const int node = (int)((blockIdx.x * blockDim.x + threadIdx.x) >> 5);
  const int lane = threadIdx.x & 31;
  if (node >= n) return;  // wave-uniform
  const int beg = rowptr[node], end = rowptr[node + 1];
  const size_t base = (size_t)node * 64;
  float2 acc = *(const float2*)(Hs + base + lane * 2);  // self-loop term
  int e = beg;
  for (; e + 2 <= end; e += 2) {
    const int s0 = srcidx[e];
    const int s1 = srcidx[e + 1];
    const float2 a0 = *(const float2*)(Hs + (size_t)s0 * 64 + lane * 2);
    const float2 a1 = *(const float2*)(Hs + (size_t)s1 * 64 + lane * 2);
    acc.x += a0.x + a1.x;
    acc.y += a0.y + a1.y;
  }
  if (e < end) {
    const int s = srcidx[e];
    const float2 v = *(const float2*)(Hs + (size_t)s * 64 + lane * 2);
    acc.x += v.x;
    acc.y += v.y;
  }
  const float di = dinv[node];
  const float v0 = acc.x * di + bias[lane * 2];
  const float v1 = acc.y * di + bias[lane * 2 + 1];
  float m = fmaxf(v0, v1);
#pragma unroll
  for (int off = 16; off > 0; off >>= 1) m = fmaxf(m, __shfl_xor(m, off, 32));
  float s = __expf(v0 - m) + __expf(v1 - m);
#pragma unroll
  for (int off = 16; off > 0; off >>= 1) s += __shfl_xor(s, off, 32);
  const float lse = m + __logf(s);
  out[base + lane * 2] = v0 - lse;
  out[base + lane * 2 + 1] = v1 - lse;
}

// ==================== launcher ====================

extern "C" void kernel_launch(void* const* d_in, const int* in_sizes, int n_in,
                              void* d_out, int out_size, void* d_ws, size_t ws_size,
                              hipStream_t stream) {
  (void)n_in; (void)out_size; (void)ws_size;

  const float* x      = (const float*)d_in[0];
  const long long* ei = (const long long*)d_in[1];  // int64 edge_index [2,E]
  const float* W1     = (const float*)d_in[2];
  const float* b1     = (const float*)d_in[3];
  const float* W2     = (const float*)d_in[4];
  const float* b2     = (const float*)d_in[5];
  float* out          = (float*)d_out;

  const int DIN = 256, DH = 256, DC = 64;
  const int N = in_sizes[0] / DIN;  // 100000 (multiple of 16 and 4)
  const int E = in_sizes[1] / 2;    // 1600000
  const long long* src = ei;
  const long long* dst = ei + E;

  // workspace layout (512B aligned slots)
  char* ws = (char*)d_ws;
  size_t off = 0;
  auto alloc = [&](size_t bytes) {
    void* p = ws + off;
    off += (bytes + 511) & ~(size_t)511;
    return p;
  };
  float* dinv      = (float*)alloc((size_t)N * 4);
  int*   rowptr    = (int*)alloc((size_t)(N + 1) * 4);
  int*   cursor    = (int*)alloc((size_t)N * 4);        // doubles as cnt
  int*   blockSums = (int*)alloc(128 * 4);
  int*   sortedSrc = (int*)alloc((size_t)E * 4);
  float* bufA      = (float*)alloc((size_t)N * DH * 4);  // Hs1, later Hs2
  float* bufC      = (float*)alloc((size_t)N * DH * 4);  // h

  const int TB = 256;
  const int G = (N + 1023) / 1024;  // scan chunks (98 <= 128)

  // ---- CSR by dst + dinv ----
  hipMemsetAsync(cursor, 0, (size_t)N * 4, stream);
  k_count<<<(E + TB - 1) / TB, TB, 0, stream>>>(dst, cursor, E);
  k_scan_partial<<<G, TB, 0, stream>>>(cursor, blockSums, N);
  k_scan_bases<<<1, 128, 0, stream>>>(blockSums, rowptr, G, N);
  k_scan_final<<<G, TB, 0, stream>>>(cursor, blockSums, rowptr, N);
  hipMemcpyAsync(cursor, rowptr, (size_t)N * 4, hipMemcpyDeviceToDevice, stream);
  k_fill<<<(E + TB - 1) / TB, TB, 0, stream>>>(src, dst, cursor, sortedSrc, E);
  k_dinv<<<(N + TB - 1) / TB, TB, 0, stream>>>(rowptr, dinv, N);

  // ---- layer 1: Hs1 = (x@W1)*dinv ; h = relu(gather-sum + bias) ----
  {
    const int mtiles = N / 16;
    const unsigned blocks = (unsigned)(((mtiles + 7) / 8) * (DH / 64));
    k_gemm_scale<<<blocks, TB, 0, stream>>>(x, W1, dinv, bufA, N, DIN, DH);
  }
  k_agg_relu<<<(N + 3) / 4, TB, 0, stream>>>(bufA, rowptr, sortedSrc, dinv, b1, bufC, N);

  // ---- layer 2: Hs2 = (h@W2)*dinv ; out = log_softmax(gather-sum + bias) ----
  {
    const int mtiles = N / 16;
    const unsigned blocks = (unsigned)(((mtiles + 7) / 8) * (DC / 64));
    k_gemm_scale<<<blocks, TB, 0, stream>>>(bufC, W2, dinv, bufA, N, DH, DC);
  }
  {
    const int waves_per_block = TB / 32;
    k_agg_logsoftmax<<<(N + waves_per_block - 1) / waves_per_block, TB, 0, stream>>>(
        bufA, rowptr, sortedSrc, dinv, b2, out, N);
  }
}
